// RUMnet_46849503265134
// MI455X (gfx1250) — compile-verified, matrix operand order checked by reading, not run on hardware
//
#include <hip/hip_runtime.h>
#include <hip/hip_bf16.h>

typedef __attribute__((ext_vector_type(16))) _Float16 v16h;
typedef __attribute__((ext_vector_type(8)))  _Float16 v8h;
typedef __attribute__((ext_vector_type(8)))  float    v8f;
typedef __attribute__((ext_vector_type(4)))  unsigned int v4u;
typedef __attribute__((__vector_size__(4 * sizeof(int)))) int v4i_t;

#define DEV __device__ __forceinline__

// ---------------- CDNA5 async global<->LDS copy helpers (guarded) ----------------
#if __has_builtin(__builtin_amdgcn_global_load_async_to_lds_b128)
#define HAVE_ASYNC_LD 1
#endif
#if __has_builtin(__builtin_amdgcn_global_store_async_from_lds_b128)
#define HAVE_ASYNC_ST 1
#endif

DEV void cp16_g2l(_Float16* lds, const _Float16* g) {
#ifdef HAVE_ASYNC_LD
    __builtin_amdgcn_global_load_async_to_lds_b128(
        (__attribute__((address_space(1))) v4i_t*)(size_t)g,
        (__attribute__((address_space(3))) v4i_t*)(unsigned)(size_t)lds, 0, 0);
#else
    *(v4u*)lds = *(const v4u*)g;
#endif
}

DEV void cp16_l2g(_Float16* g, const _Float16* lds) {
#ifdef HAVE_ASYNC_ST
    __builtin_amdgcn_global_store_async_from_lds_b128(
        (__attribute__((address_space(1))) v4i_t*)(size_t)g,
        (__attribute__((address_space(3))) v4i_t*)(unsigned)(size_t)lds, 0, 0);
#else
    *(v4u*)g = *(const v4u*)lds;
#endif
}

DEV void wait_async() {
#if __has_builtin(__builtin_amdgcn_s_wait_asynccnt)
    __builtin_amdgcn_s_wait_asynccnt(0);
#elif defined(HAVE_ASYNC_LD) || defined(HAVE_ASYNC_ST)
    asm volatile("s_wait_asynccnt 0x0" ::: "memory");
#endif
    asm volatile("" ::: "memory");
}

DEV void wait_ds() {
#if __has_builtin(__builtin_amdgcn_s_wait_dscnt)
    __builtin_amdgcn_s_wait_dscnt(0);
#else
    asm volatile("s_wait_dscnt 0x0" ::: "memory");
#endif
    asm volatile("" ::: "memory");
}

// ---------------- WMMA fragment load (CDNA5 16-bit 16x32 A/B layout) ----------------
//   lanes 0-15 : row = lane,    K = ks*32 + {0..7, 16..23}
//   lanes 16-31: row = lane-16, K = ks*32 + {8..15, 24..31}
DEV v16h load_frag(const _Float16* base, int pitch, int kstep) {
    const int lane = threadIdx.x & 31;
    const int mi   = lane & 15;
    const int k0   = kstep * 32 + ((lane >> 4) << 3);
    const _Float16* p = base + mi * pitch + k0;
    v8h lo = *(const v8h*)(p);
    v8h hi = *(const v8h*)(p + 16);
    return __builtin_shufflevector(lo, hi, 0,1,2,3,4,5,6,7,8,9,10,11,12,13,14,15);
}

template<int NT, int KS>
DEV void gemm_tile(const _Float16* A, int pa, const _Float16* W, int pw, v8f acc[NT]) {
#pragma unroll
    for (int nt = 0; nt < NT; ++nt)
        acc[nt] = (v8f){0.f,0.f,0.f,0.f,0.f,0.f,0.f,0.f};
#pragma unroll
    for (int ks = 0; ks < KS; ++ks) {
        v16h a = load_frag(A, pa, ks);
#pragma unroll
        for (int nt = 0; nt < NT; ++nt) {
            v16h b = load_frag(W + nt * 16 * pw, pw, ks);
            acc[nt] = __builtin_amdgcn_wmma_f32_16x16x32_f16(
                false, a, false, b, (short)0, acc[nt], false, false);
        }
    }
}

DEV float elu(float x) { return x > 0.f ? x : __expf(x) - 1.f; }

// C/D layout: lane -> column (lane&15)+16*nt ; VGPR v -> row v + 8*(lane>>4)
template<int NT>
DEV void bias_elu_store(v8f acc[NT], const float* bias, _Float16* out, int pitch) {
    const int lane = threadIdx.x & 31;
    const int col  = lane & 15;
    const int rh   = (lane >> 4) * 8;
#pragma unroll
    for (int nt = 0; nt < NT; ++nt) {
        float bv = bias[nt * 16 + col];
#pragma unroll
        for (int v = 0; v < 8; ++v) {
            float x = elu(acc[nt][v] + bv);
            out[(rh + v) * pitch + nt * 16 + col] = (_Float16)x;
        }
    }
}

#define PK1 72    // eps/nu LDS pitch (halves): K 64 + 8 skew
#define PU0 264   // utility layer-0 pitch: K 232 padded to 256 + 8 skew
#define PUR 136   // utility later-layer pitch: K 128 + 8 skew

// ============ Kernel 0: one-time f32->f16 conversion + weight transpose/pad ============
// Segments (element counts): X16 204800 | Z16 16384 | WuT0 33792 | WuTR 69632 |
//                            WxT0 46080 | WxTR 184320 | WzT0 46080 | WzTR 184320
__global__ __launch_bounds__(256) void k_prep(
    const float* __restrict__ X, const float* __restrict__ Z,
    const float* __restrict__ Wx0, const float* __restrict__ WxR,
    const float* __restrict__ Wz0, const float* __restrict__ WzR,
    const float* __restrict__ Wu0, const float* __restrict__ WuR,
    _Float16* __restrict__ X16, _Float16* __restrict__ Z16,
    _Float16* __restrict__ WuT0g, _Float16* __restrict__ WuTRg,
    _Float16* __restrict__ WxT0g, _Float16* __restrict__ WxTRg,
    _Float16* __restrict__ WzT0g, _Float16* __restrict__ WzTRg) {
    int id = blockIdx.x * 256 + threadIdx.x;
    if (id < 204800) { X16[id] = (_Float16)X[id]; return; }
    id -= 204800;
    if (id < 16384) { Z16[id] = (_Float16)Z[id]; return; }
    id -= 16384;
    if (id < 128 * PU0) {
        int n = id / PU0, k = id % PU0;
        WuT0g[id] = (k < 232) ? (_Float16)Wu0[k * 128 + n] : (_Float16)0.f; return;
    }
    id -= 128 * PU0;
    if (id < 4 * 128 * PUR) {
        int l = id / (128 * PUR), r = id % (128 * PUR), n = r / PUR, k = r % PUR;
        WuTRg[id] = (k < 128) ? (_Float16)WuR[l * 16384 + k * 128 + n] : (_Float16)0.f; return;
    }
    id -= 4 * 128 * PUR;
    if (id < 10 * 64 * PK1) {
        int h = id / (64 * PK1), r = id % (64 * PK1), n = r / PK1, k = r % PK1;
        WxT0g[id] = (k < 40) ? (_Float16)Wx0[h * 2560 + k * 64 + n] : (_Float16)0.f; return;
    }
    id -= 10 * 64 * PK1;
    if (id < 40 * 64 * PK1) {
        int lh = id / (64 * PK1), r = id % (64 * PK1), n = r / PK1, k = r % PK1;
        WxTRg[id] = (k < 64) ? (_Float16)WxR[lh * 4096 + k * 64 + n] : (_Float16)0.f; return;
    }
    id -= 40 * 64 * PK1;
    if (id < 10 * 64 * PK1) {
        int h = id / (64 * PK1), r = id % (64 * PK1), n = r / PK1, k = r % PK1;
        WzT0g[id] = (k < 64) ? (_Float16)Wz0[h * 4096 + k * 64 + n] : (_Float16)0.f; return;
    }
    id -= 10 * 64 * PK1;
    if (id < 40 * 64 * PK1) {
        int lh = id / (64 * PK1), r = id % (64 * PK1), n = r / PK1, k = r % PK1;
        WzTRg[id] = (k < 64) ? (_Float16)WzR[lh * 4096 + k * 64 + n] : (_Float16)0.f; return;
    }
}

// ============ Kernel 1: epsilon(X) nets, 10 heads x 5 layers; rows=5120, N=64 ==========
__global__ __launch_bounds__(128) void k_xh(const _Float16* __restrict__ X16,
                                            const _Float16* __restrict__ WxT0g,
                                            const _Float16* __restrict__ WxTRg,
                                            const float* __restrict__ bx,
                                            _Float16* __restrict__ XhF) {
    __shared__ __align__(16) _Float16 sW[64 * PK1];
    __shared__ __align__(16) _Float16 sAct[4][16 * PK1];
    const int h = blockIdx.x / 80, wg = blockIdx.x % 80;
    const int wave = threadIdx.x >> 5, lane = threadIdx.x & 31;
    const int r0 = (wg * 4 + wave) * 16;
    _Float16* act = sAct[wave];

    for (int idx = lane; idx < 80; idx += 32) {            // stage 16 rows x 40 halves
        int m = idx / 5, c = idx % 5;
        cp16_g2l(act + m * PK1 + c * 8, X16 + (r0 + m) * 40 + c * 8);
    }
    for (int idx = lane; idx < 48; idx += 32) {            // zero pad K 40..63
        int m = idx / 3, c = idx % 3;
        *(v4u*)(act + m * PK1 + 40 + c * 8) = (v4u){0u,0u,0u,0u};
    }
    for (int l = 0; l < 5; ++l) {
        __syncthreads();
        const _Float16* wsrc = (l == 0) ? (WxT0g + h * 64 * PK1)
                                        : (WxTRg + ((l - 1) * 10 + h) * 64 * PK1);
        for (int idx = threadIdx.x; idx < 64 * PK1 / 8; idx += 128)
            cp16_g2l(sW + idx * 8, wsrc + idx * 8);
        wait_async();
        __syncthreads();
        v8f acc[4];
        gemm_tile<4, 2>(act, PK1, sW, PK1, acc);
        bias_elu_store<4>(acc, bx + (l * 10 + h) * 64, act, PK1);
    }
    wait_ds();                                             // epilogue ds stores -> LDS retired
    for (int idx = lane; idx < 128; idx += 32) {           // async store 16x64 f16 out
        int m = idx >> 3, c = idx & 7;
        cp16_l2g(XhF + (h * 5120 + r0 + m) * 64 + c * 8, act + m * PK1 + c * 8);
    }
}

// ============ Kernel 2: nu(Z) nets, 10 heads x 5 layers; rows=256, N=64 ================
__global__ __launch_bounds__(128) void k_zh(const _Float16* __restrict__ Z16,
                                            const _Float16* __restrict__ WzT0g,
                                            const _Float16* __restrict__ WzTRg,
                                            const float* __restrict__ bz,
                                            _Float16* __restrict__ zhF) {
    __shared__ __align__(16) _Float16 sW[64 * PK1];
    __shared__ __align__(16) _Float16 sAct[4][16 * PK1];
    const int h = blockIdx.x / 4, wg = blockIdx.x % 4;
    const int wave = threadIdx.x >> 5, lane = threadIdx.x & 31;
    const int r0 = (wg * 4 + wave) * 16;
    _Float16* act = sAct[wave];

    for (int idx = lane; idx < 128; idx += 32) {
        int m = idx >> 3, c = idx & 7;
        cp16_g2l(act + m * PK1 + c * 8, Z16 + (r0 + m) * 64 + c * 8);
    }
    for (int l = 0; l < 5; ++l) {
        __syncthreads();
        const _Float16* wsrc = (l == 0) ? (WzT0g + h * 64 * PK1)
                                        : (WzTRg + ((l - 1) * 10 + h) * 64 * PK1);
        for (int idx = threadIdx.x; idx < 64 * PK1 / 8; idx += 128)
            cp16_g2l(sW + idx * 8, wsrc + idx * 8);
        wait_async();
        __syncthreads();
        v8f acc[4];
        gemm_tile<4, 2>(act, PK1, sW, PK1, acc);
        bias_elu_store<4>(acc, bz + (l * 10 + h) * 64, act, PK1);
    }
    wait_ds();
    for (int idx = lane; idx < 128; idx += 32) {
        int m = idx >> 3, c = idx & 7;
        cp16_l2g(zhF + (h * 256 + r0 + m) * 64 + c * 8, act + m * PK1 + c * 8);
    }
}

// ============ Kernel 3: shared utility MLP over 100 (hx,hz) pairs ======================
// grid = 100*10 WGs, 128 thr; each wave does 8 tiles of 16 rows. L0 weights LDS-resident.
__global__ __launch_bounds__(128) void k_util(const _Float16* __restrict__ X16,
                                              const _Float16* __restrict__ Z16,
                                              const _Float16* __restrict__ XhF,
                                              const _Float16* __restrict__ zhF,
                                              const _Float16* __restrict__ WuT0g,
                                              const _Float16* __restrict__ WuTRg,
                                              const float* __restrict__ bu,
                                              const float* __restrict__ Wlast,
                                              float* __restrict__ logits) {
    extern __shared__ __align__(16) _Float16 smem[];
    _Float16* sW0  = smem;                            // 128*PU0 (resident layer-0 W^T)
    _Float16* sWr  = smem + 128 * PU0;                // 128*PUR (streamed layer 1..4 W^T)
    _Float16* sAct = smem + 128 * PU0 + 128 * PUR;    // 4 waves * 16*PU0
    const int p = blockIdx.x / 10, chunk = blockIdx.x % 10;
    const int hx = p / 10, hz = p % 10;
    const int wave = threadIdx.x >> 5, lane = threadIdx.x & 31;
    _Float16* act = sAct + wave * 16 * PU0;

    for (int idx = threadIdx.x; idx < 128 * PU0 / 8; idx += 128)   // L0 weights, once
        cp16_g2l(sW0 + idx * 8, WuT0g + idx * 8);

    for (int it = 0; it < 8; ++it) {
        const int r0 = (chunk * 32 + it * 4 + wave) * 16;
        // ---- stage concat([X(40) | Xh(64) | zh(64) | Z(64) | pad(24)]) via async DMA
        for (int idx = lane; idx < 80; idx += 32) {                // X
            int m = idx / 5, c = idx % 5;
            cp16_g2l(act + m * PU0 + c * 8, X16 + (r0 + m) * 40 + c * 8);
        }
        for (int idx = lane; idx < 128; idx += 32) {               // Xh
            int m = idx >> 3, c = idx & 7;
            cp16_g2l(act + m * PU0 + 40 + c * 8, XhF + (hx * 5120 + r0 + m) * 64 + c * 8);
        }
        for (int idx = lane; idx < 128; idx += 32) {               // zh
            int m = idx >> 3, c = idx & 7;
            cp16_g2l(act + m * PU0 + 104 + c * 8, zhF + (hz * 256 + (r0 + m) / 20) * 64 + c * 8);
        }
        for (int idx = lane; idx < 128; idx += 32) {               // Z
            int m = idx >> 3, c = idx & 7;
            cp16_g2l(act + m * PU0 + 168 + c * 8, Z16 + ((r0 + m) / 20) * 64 + c * 8);
        }
        for (int idx = lane; idx < 48; idx += 32) {                // pad 232..255 zero
            int m = idx / 3, c = idx % 3;
            *(v4u*)(act + m * PU0 + 232 + c * 8) = (v4u){0u,0u,0u,0u};
        }
        wait_async();
        if (it == 0) __syncthreads();     // sW0 visible to all waves
        // ---- layer 0: 232(pad 256) -> 128
        v8f acc[8];
        gemm_tile<8, 8>(act, PU0, sW0, PU0, acc);
        bias_elu_store<8>(acc, bu, act, PUR);
        // ---- layers 1..4: 128 -> 128 (weights streamed into sWr)
        for (int l = 1; l < 5; ++l) {
            __syncthreads();
            {
                const _Float16* wsrc = WuTRg + (l - 1) * 128 * PUR;
                for (int idx = threadIdx.x; idx < 128 * PUR / 8; idx += 128)
                    cp16_g2l(sWr + idx * 8, wsrc + idx * 8);
            }
            wait_async();
            __syncthreads();
            gemm_tile<8, 4>(act, PUR, sWr, PUR, acc);
            if (l < 4) {
                bias_elu_store<8>(acc, bu + l * 128, act, PUR);
            } else {
                const int col = lane & 15;
                const int rh  = (lane >> 4) * 8;
                float part[8];
#pragma unroll
                for (int v = 0; v < 8; ++v) part[v] = 0.f;
#pragma unroll
                for (int nt = 0; nt < 8; ++nt) {
                    float bv = bu[4 * 128 + nt * 16 + col];
                    float wl = Wlast[nt * 16 + col];
#pragma unroll
                    for (int v = 0; v < 8; ++v)
                        part[v] += elu(acc[nt][v] + bv) * wl;
                }
#pragma unroll
                for (int v = 0; v < 8; ++v) {
                    part[v] += __shfl_xor(part[v], 1);
                    part[v] += __shfl_xor(part[v], 2);
                    part[v] += __shfl_xor(part[v], 4);
                    part[v] += __shfl_xor(part[v], 8);
                }
                if (col == 0) {
#pragma unroll
                    for (int v = 0; v < 8; ++v)
                        logits[p * 5120 + r0 + rh + v] = part[v];
                }
            }
        }
    }
}

// ============ Kernel 4: softmax over A=20, mean over 100 pairs =========================
__global__ __launch_bounds__(32) void k_softmax(const float* __restrict__ logits,
                                                float* __restrict__ out) {
    const int b = blockIdx.x * 32 + threadIdx.x;   // 0..255
    float acc[20];
#pragma unroll
    for (int a = 0; a < 20; ++a) acc[a] = 0.f;
    for (int p = 0; p < 100; ++p) {
        const float* l = logits + p * 5120 + b * 20;
        float lv[20], mx = -1e30f;
#pragma unroll
        for (int a = 0; a < 20; ++a) { lv[a] = l[a]; mx = fmaxf(mx, lv[a]); }
        float s = 0.f;
#pragma unroll
        for (int a = 0; a < 20; ++a) { lv[a] = __expf(lv[a] - mx); s += lv[a]; }
        float inv = 1.f / s;
#pragma unroll
        for (int a = 0; a < 20; ++a) acc[a] += lv[a] * inv;
    }
    const float TOLc = 1e-6f;
#pragma unroll
    for (int a = 0; a < 20; ++a)
        out[b * 20 + a] = (TOLc + acc[a] * 0.01f) / (1.f + TOLc * 20.f);
}

extern "C" void kernel_launch(void* const* d_in, const int* in_sizes, int n_in,
                              void* d_out, int out_size, void* d_ws, size_t ws_size,
                              hipStream_t stream) {
    (void)in_sizes; (void)n_in; (void)out_size; (void)ws_size;
    const float* X     = (const float*)d_in[0];
    const float* Z     = (const float*)d_in[1];
    const float* Wx0   = (const float*)d_in[2];
    const float* WxR   = (const float*)d_in[3];
    const float* bx    = (const float*)d_in[4];
    const float* Wz0   = (const float*)d_in[5];
    const float* WzR   = (const float*)d_in[6];
    const float* bz    = (const float*)d_in[7];
    const float* Wu0   = (const float*)d_in[8];
    const float* WuR   = (const float*)d_in[9];
    const float* bu    = (const float*)d_in[10];
    const float* Wlast = (const float*)d_in[11];
    float* out = (float*)d_out;

    // workspace layout (bytes, 16B aligned)
    char* ws = (char*)d_ws;
    _Float16* XhF   = (_Float16*)(ws);                        // 10*5120*64 f16
    _Float16* zhF   = (_Float16*)(ws + 6553600);              // 10*256*64  f16
    float*    logit = (float*)(ws + 6881280);                 // 100*5120   f32
    _Float16* X16   = (_Float16*)(ws + 8929280);              // 5120*40    f16
    _Float16* Z16   = (_Float16*)(ws + 9338880);              // 256*64     f16
    _Float16* WuT0g = (_Float16*)(ws + 9371648);              // 128*264    f16
    _Float16* WuTRg = (_Float16*)(ws + 9439232);              // 4*128*136  f16
    _Float16* WxT0g = (_Float16*)(ws + 9578496);              // 10*64*72   f16
    _Float16* WxTRg = (_Float16*)(ws + 9670656);              // 40*64*72   f16
    _Float16* WzT0g = (_Float16*)(ws + 10039296);             // 10*64*72   f16
    _Float16* WzTRg = (_Float16*)(ws + 10131456);             // 40*64*72   f16

    const int dynLds = (128 * PU0 + 128 * PUR + 4 * 16 * PU0) * (int)sizeof(_Float16); // 136192
    (void)hipFuncSetAttribute((const void*)k_util,
                              hipFuncAttributeMaxDynamicSharedMemorySize, dynLds);

    k_prep<<<3068, 256, 0, stream>>>(X, Z, Wx0, WxR, Wz0, WzR, Wu0, WuR,
                                     X16, Z16, WuT0g, WuTRg, WxT0g, WxTRg, WzT0g, WzTRg);
    k_xh<<<800, 128, 0, stream>>>(X16, WxT0g, WxTRg, bx, XhF);
    k_zh<<<40, 128, 0, stream>>>(Z16, WzT0g, WzTRg, bz, zhF);
    k_util<<<1000, 128, dynLds, stream>>>(X16, Z16, XhF, zhF, WuT0g, WuTRg, bu, Wlast, logit);
    k_softmax<<<8, 32, 0, stream>>>(logit, out);
}
